// TransformerCache_42245298324242
// MI455X (gfx1250) — compile-verified, matrix operand order checked by reading, not run on hardware
//
#include <hip/hip_runtime.h>
#include <hip/hip_bf16.h>

#define B_      2
#define S_      1024
#define H_      32
#define D_      128
#define MAXSEQ_ 4096
#define START_  1024
#define KT_     (START_ + S_)   // 2048 total keys
#define D2_     (D_ / 2)

typedef __attribute__((ext_vector_type(16))) _Float16 v16h;
typedef __attribute__((ext_vector_type(8)))  _Float16 v8h;
typedef __attribute__((ext_vector_type(8)))  float    v8f;

typedef unsigned int u32;
typedef __attribute__((ext_vector_type(4))) unsigned int u32x4;
typedef __attribute__((ext_vector_type(8))) int          i32x8;
typedef __attribute__((ext_vector_type(4))) int          i32x4;

#if defined(__HIP_DEVICE_COMPILE__) && __has_builtin(__builtin_amdgcn_tensor_load_to_lds)
#define USE_TDM 1
#else
#define USE_TDM 0
#endif

__device__ __forceinline__ v16h cat16(v8h lo, v8h hi) {
  return __builtin_shufflevector(lo, hi, 0,1,2,3,4,5,6,7,8,9,10,11,12,13,14,15);
}

#if USE_TDM
__device__ __forceinline__ void tdm_wait0() {
#if __has_builtin(__builtin_amdgcn_s_wait_tensorcnt)
  __builtin_amdgcn_s_wait_tensorcnt((short)0);
#else
  asm volatile("s_wait_tensorcnt 0" ::: "memory");
#endif
}

// 2D tile load via Tensor Data Mover. Dims in 2-byte elements.
// pad_iv: pad after 2<<pad_iv dwords; pad_amt: (pad_amt+1) dwords of LDS pad.
__device__ __forceinline__ void tdm_load_2d(u32 lds_addr, const _Float16* gptr,
                                            int tensor_d0, int tensor_d1,
                                            int tile_d0, int tile_d1,
                                            int stride0, int pad_iv, int pad_amt) {
  unsigned long long ga = (unsigned long long)(uintptr_t)gptr;
  u32x4 g0;
  g0.x = 1u;                                    // count=1 (valid user D#)
  g0.y = lds_addr;                              // LDS byte address
  g0.z = (u32)(ga & 0xffffffffu);               // global_addr[31:0]
  g0.w = (u32)(ga >> 32) | (2u << 30);          // global_addr[56:32] | type=2
  i32x8 g1;
  g1[0] = (1 << 16) | (1 << 20) | (pad_iv << 22) | (pad_amt << 25); // 2B data, pad_enable
  g1[1] = (tensor_d0 & 0xffff) << 16;                               // tensor_dim0 lo
  g1[2] = ((tensor_d0 >> 16) & 0xffff) | ((tensor_d1 & 0xffff) << 16);
  g1[3] = ((tensor_d1 >> 16) & 0xffff) | ((tile_d0 & 0xffff) << 16);
  g1[4] = (tile_d1 & 0xffff);                                       // tile_dim1, tile_dim2=0
  g1[5] = stride0;                                                  // tensor_dim0_stride lo
  g1[6] = 0;
  g1[7] = 0;
  i32x4 gz = {0, 0, 0, 0};
#if __has_include(<hip/amd_detail/amd_gfx1250_TDM.h>)
  i32x8 gz8 = {0, 0, 0, 0, 0, 0, 0, 0};
  __builtin_amdgcn_tensor_load_to_lds(g0, g1, gz, gz, gz8, 0);
#else
  __builtin_amdgcn_tensor_load_to_lds(g0, g1, gz, gz, 0);
#endif
}
#endif  // USE_TDM

// --- Prep 1: RoPE(q) * 1/sqrt(D), f32 -> f16, layout (B,S,H,D) -> (B,H,S,D)
__global__ void prep_q_kernel(const float* __restrict__ q,
                              const float* __restrict__ cs,
                              const float* __restrict__ sn,
                              _Float16* __restrict__ Qf) {
  int idx = blockIdx.x * blockDim.x + threadIdx.x;       // over B*S*H*D2
  if (idx >= B_ * S_ * H_ * D2_) return;
  int d2 = idx % D2_;
  int h  = (idx / D2_) % H_;
  int s  = (idx / (D2_ * H_)) % S_;
  int b  =  idx / (D2_ * H_ * S_);
  float c  = cs[s * D2_ + d2];
  float si = sn[s * D2_ + d2];
  size_t src = (((size_t)b * S_ + s) * H_ + h) * D_ + 2 * d2;
  float x0 = q[src], x1 = q[src + 1];
  const float scale = 0.08838834764831845f;              // 1/sqrt(128)
  size_t dst = (((size_t)b * H_ + h) * S_ + s) * D_ + 2 * d2;
  Qf[dst]     = (_Float16)((x0 * c - x1 * si) * scale);
  Qf[dst + 1] = (_Float16)((x1 * c + x0 * si) * scale);
}

// --- Prep 2: full key tensor [cache_k ; rope(k)] -> f16 (B,H,KT,D)
__global__ void prep_k_kernel(const float* __restrict__ k,
                              const float* __restrict__ cache_k,
                              const float* __restrict__ cs,
                              const float* __restrict__ sn,
                              _Float16* __restrict__ Kf) {
  int idx = blockIdx.x * blockDim.x + threadIdx.x;       // over B*H*KT*D2
  if (idx >= B_ * H_ * KT_ * D2_) return;
  int d2 = idx % D2_;
  int t  = (idx / D2_) % KT_;
  int h  = (idx / (D2_ * KT_)) % H_;
  int b  =  idx / (D2_ * KT_ * H_);
  size_t dst = (((size_t)b * H_ + h) * KT_ + t) * D_ + 2 * d2;
  if (t < START_) {
    size_t src = (((size_t)b * MAXSEQ_ + t) * H_ + h) * D_ + 2 * d2;
    Kf[dst]     = (_Float16)cache_k[src];
    Kf[dst + 1] = (_Float16)cache_k[src + 1];
  } else {
    int s = t - START_;
    size_t src = (((size_t)b * S_ + s) * H_ + h) * D_ + 2 * d2;
    float c  = cs[s * D2_ + d2];
    float si = sn[s * D2_ + d2];
    float x0 = k[src], x1 = k[src + 1];
    Kf[dst]     = (_Float16)(x0 * c - x1 * si);
    Kf[dst + 1] = (_Float16)(x1 * c + x0 * si);
  }
}

// --- Prep 3: full value tensor transposed -> f16 (B,H,D,KT)
__global__ void prep_vt_kernel(const float* __restrict__ v,
                               const float* __restrict__ cache_v,
                               _Float16* __restrict__ VT) {
  int idx = blockIdx.x * blockDim.x + threadIdx.x;       // over B*H*D*KT
  if (idx >= B_ * H_ * D_ * KT_) return;
  int t = idx % KT_;
  int d = (idx / KT_) % D_;
  int h = (idx / (KT_ * D_)) % H_;
  int b =  idx / (KT_ * D_ * H_);
  float val;
  if (t < START_)
    val = cache_v[(((size_t)b * MAXSEQ_ + t) * H_ + h) * D_ + d];
  else
    val = v[(((size_t)b * S_ + (t - START_)) * H_ + h) * D_ + d];
  VT[(((size_t)b * H_ + h) * D_ + d) * KT_ + t] = (_Float16)val;
}

// --- Flash attention: 8 waves/block, one 16-query tile per wave, 32 keys/step.
// K/V tiles staged to LDS once per block (TDM, double-buffered).
__global__ __launch_bounds__(256) void attn_kernel(const _Float16* __restrict__ Qf,
                                                   const _Float16* __restrict__ Kf,
                                                   const _Float16* __restrict__ VT,
                                                   float* __restrict__ out) {
  constexpr int KROW = D_ + 8;   // 136 halves: 272B stride -> conflict-free b128
  constexpr int VROW = 32 + 8;   // 40 halves:  80B stride -> conflict-free b128
  constexpr int PROW = 32 + 8;
  __shared__ __align__(16) _Float16 kbuf[2][32 * KROW];
  __shared__ __align__(16) _Float16 vbuf[2][D_ * VROW];
  __shared__ __align__(16) _Float16 pbuf[8][16 * PROW];

  const int lane   = threadIdx.x & 31;
  const int wave   = threadIdx.x >> 5;
  const int half16 = lane >> 4;
  const int l16    = lane & 15;

  const int qblocks = S_ / 128;
  const int bh    = blockIdx.x / qblocks;
  const int qblk  = blockIdx.x % qblocks;
  const int b     = bh / H_;
  const int h     = bh % H_;
  const int qbase = qblk * 128 + wave * 16;

  // A fragments: Q tile 16x128 as 4 x (16x32) f16
  const _Float16* qp = Qf + ((size_t)bh * S_ + qbase + l16) * D_;
  v16h aq[4];
#pragma unroll
  for (int c = 0; c < 4; ++c) {
    const int d0 = c * 32 + half16 * 8;
    aq[c] = cat16(*(const v8h*)(qp + d0), *(const v8h*)(qp + d0 + 16));
  }

  const _Float16* kp  = Kf + (size_t)bh * KT_ * D_;
  const _Float16* vtp = VT + (size_t)bh * D_ * KT_;

  const v8f vzero = {0.f, 0.f, 0.f, 0.f, 0.f, 0.f, 0.f, 0.f};
  v8f ot[8];
#pragma unroll
  for (int t = 0; t < 8; ++t) ot[t] = vzero;
  float mrow[8], lrow[8];
#pragma unroll
  for (int j = 0; j < 8; ++j) { mrow[j] = -1e30f; lrow[j] = 0.f; }

  const int qpos_lo = START_ + qbase;
  const int ntiles  = (qpos_lo + 16 + 31) >> 5;                    // this wave
  const int ntmax   = (START_ + qblk * 128 + 112 + 16 + 31) >> 5;  // whole block

#if USE_TDM
  if (wave == 0) {
    tdm_load_2d((u32)(uintptr_t)&kbuf[0][0], kp,  D_,  KT_, D_, 32, D_,  5, 3);
    tdm_load_2d((u32)(uintptr_t)&vbuf[0][0], vtp, KT_, D_,  32, D_, KT_, 3, 3);
  }
#endif

  for (int kt = 0; kt < ntmax; ++kt) {
    const _Float16* kT;
    const _Float16* vT;
#if USE_TDM
    if (wave == 0) tdm_wait0();               // tile kt resident in buf[kt&1]
    asm volatile("" ::: "memory");
    __syncthreads();
    if (wave == 0 && kt + 1 < ntmax) {        // prior readers of buf done: refill
      const int nb = (kt + 1) & 1;
      const int kb1 = (kt + 1) << 5;
      tdm_load_2d((u32)(uintptr_t)&kbuf[nb][0], kp + (size_t)kb1 * D_,
                  D_, KT_, D_, 32, D_, 5, 3);
      tdm_load_2d((u32)(uintptr_t)&vbuf[nb][0], vtp + (size_t)kb1,
                  KT_, D_, 32, D_, KT_, 3, 3);
    }
    kT = &kbuf[kt & 1][0];
    vT = &vbuf[kt & 1][0];
#else
    __syncthreads();
    {                                          // cooperative synchronous staging
      const int kb0 = kt << 5;
#pragma unroll
      for (int i = 0; i < 2; ++i) {
        const int c = threadIdx.x * 2 + i;     // 0..511 16B chunks
        const int r = c >> 4, off = (c & 15) * 8;
        *(v8h*)&kbuf[0][r * KROW + off] =
            *(const v8h*)(kp + (size_t)(kb0 + r) * D_ + off);
        const int dr = c >> 2, doff = (c & 3) * 8;
        *(v8h*)&vbuf[0][dr * VROW + doff] =
            *(const v8h*)(vtp + (size_t)dr * KT_ + kb0 + doff);
      }
    }
    __syncthreads();
    kT = &kbuf[0][0];
    vT = &vbuf[0][0];
#endif

    if (kt < ntiles) {
      const int kb = kt << 5;

      // S = Q @ K^T : two 16-key sub-tiles, 4 k-chunks each (8 WMMA)
      v8f s0 = vzero, s1 = vzero;
#pragma unroll
      for (int c = 0; c < 4; ++c) {
        const int d0 = c * 32 + half16 * 8;
        const _Float16* kr0 = kT + l16 * KROW + d0;
        const _Float16* kr1 = kr0 + 16 * KROW;
        v16h bk0 = cat16(*(const v8h*)kr0, *(const v8h*)(kr0 + 16));
        v16h bk1 = cat16(*(const v8h*)kr1, *(const v8h*)(kr1 + 16));
        s0 = __builtin_amdgcn_wmma_f32_16x16x32_f16(false, aq[c], false, bk0,
                                                    (short)0, s0, false, false);
        s1 = __builtin_amdgcn_wmma_f32_16x16x32_f16(false, aq[c], false, bk1,
                                                    (short)0, s1, false, false);
      }

      // causal mask only on diagonal-straddling tiles
      if (kb + 31 > qpos_lo) {
#pragma unroll
        for (int j = 0; j < 8; ++j) {
          const int qposm = qpos_lo + j + 8 * half16;
          if (kb + l16 > qposm)      s0[j] = -1e30f;
          if (kb + 16 + l16 > qposm) s1[j] = -1e30f;
        }
      }

      // online softmax: per-row reductions across the 16-lane half
      _Float16* pw = &pbuf[wave][0];
      float alpha[8];
#pragma unroll
      for (int j = 0; j < 8; ++j) {
        float rm = fmaxf(s0[j], s1[j]);
        rm = fmaxf(rm, __shfl_xor(rm, 1, 32));
        rm = fmaxf(rm, __shfl_xor(rm, 2, 32));
        rm = fmaxf(rm, __shfl_xor(rm, 4, 32));
        rm = fmaxf(rm, __shfl_xor(rm, 8, 32));
        const float mo = mrow[j];
        const float mn = fmaxf(mo, rm);
        const float p0 = __expf(s0[j] - mn);
        const float p1 = __expf(s1[j] - mn);
        float rs = p0 + p1;
        rs += __shfl_xor(rs, 1, 32);
        rs += __shfl_xor(rs, 2, 32);
        rs += __shfl_xor(rs, 4, 32);
        rs += __shfl_xor(rs, 8, 32);
        const float al = __expf(mo - mn);
        mrow[j]  = mn;
        lrow[j]  = lrow[j] * al + rs;
        alpha[j] = al;
        const int m = j + 8 * half16;          // stage P (f16) row-major
        pw[m * PROW + l16]      = (_Float16)p0;
        pw[m * PROW + 16 + l16] = (_Float16)p1;
      }

      // rescale accumulator
#pragma unroll
      for (int t = 0; t < 8; ++t)
#pragma unroll
        for (int j = 0; j < 8; ++j) ot[t][j] *= alpha[j];

      // same-wave LDS RAW fence, reload P in A-fragment layout
      asm volatile("s_wait_dscnt 0" ::: "memory");
      const _Float16* pr = pw + l16 * PROW + half16 * 8;
      v16h pa = cat16(*(const v8h*)pr, *(const v8h*)(pr + 16));

      // O += P @ V : 8 output column tiles (8 WMMA)
#pragma unroll
      for (int t = 0; t < 8; ++t) {
        const _Float16* vr = vT + (t * 16 + l16) * VROW + half16 * 8;
        v16h vb = cat16(*(const v8h*)vr, *(const v8h*)(vr + 16));
        ot[t] = __builtin_amdgcn_wmma_f32_16x16x32_f16(false, pa, false, vb,
                                                       (short)0, ot[t], false, false);
      }
    }
  }

  // epilogue: out (B,S,H,D) f32, divide by row sum
  float* ob = out + (((size_t)b * S_ + qbase) * H_ + h) * D_;
#pragma unroll
  for (int j = 0; j < 8; ++j) {
    const int m = j + 8 * half16;
    const float inv = 1.0f / lrow[j];
    float* op = ob + (size_t)m * H_ * D_ + l16;
#pragma unroll
    for (int t = 0; t < 8; ++t) op[t * 16] = ot[t][j] * inv;
  }
}

extern "C" void kernel_launch(void* const* d_in, const int* in_sizes, int n_in,
                              void* d_out, int out_size, void* d_ws, size_t ws_size,
                              hipStream_t stream) {
  const float* q  = (const float*)d_in[0];
  const float* k  = (const float*)d_in[1];
  const float* v  = (const float*)d_in[2];
  const float* ck = (const float*)d_in[3];
  const float* cv = (const float*)d_in[4];
  const float* cs = (const float*)d_in[5];
  const float* sn = (const float*)d_in[6];
  float* out = (float*)d_out;

  _Float16* Qf = (_Float16*)d_ws;                         // (B,H,S,D)
  _Float16* Kf = Qf + (size_t)B_ * H_ * S_ * D_;          // (B,H,KT,D)
  _Float16* VT = Kf + (size_t)B_ * H_ * KT_ * D_;         // (B,H,D,KT)

  const int nq = B_ * S_ * H_ * D2_;
  prep_q_kernel<<<(nq + 255) / 256, 256, 0, stream>>>(q, cs, sn, Qf);
  const int nk = B_ * H_ * KT_ * D2_;
  prep_k_kernel<<<(nk + 255) / 256, 256, 0, stream>>>(k, ck, cs, sn, Kf);
  const int nv = B_ * H_ * D_ * KT_;
  prep_vt_kernel<<<(nv + 255) / 256, 256, 0, stream>>>(v, cv, VT);

  attn_kernel<<<B_ * H_ * (S_ / 128), 256, 0, stream>>>(Qf, Kf, VT, out);
}
